// SchNet_42399917146190
// MI455X (gfx1250) — matrix operand order repaired and checked by reading, not compiled, not run on hardware
//
#include <hip/hip_runtime.h>
#include <hip/hip_bf16.h>
#include <cstdint>
#include <math.h>

// ---------------- problem constants ----------------
#define N_ATOMS   40000
#define N_EDGES   640000
#define NB        128      // n_basis == n_filters
#define N_RBF     20
#define KPAD      32       // RBF K padded to one WMMA K-step
#define N_LAYERSC 3
#define CUTOFF_F  5.0f

typedef __bf16 bf16_t;
typedef bf16_t v16bf __attribute__((ext_vector_type(16)));
typedef float  v8f   __attribute__((ext_vector_type(8)));

// ---------------- small helpers ----------------
// native f32 -> bf16 (RNE) via compiler; returns raw u16 bits
__device__ __forceinline__ unsigned short bf1(float x) {
  union { bf16_t b; unsigned short s; } c;
  c.b = (bf16_t)x;
  return c.s;
}
__device__ __forceinline__ float bf2f(unsigned short h) {
  return __uint_as_float(((uint32_t)h) << 16);
}
// shifted softplus: softplus(x) - ln(2), numerically stable
__device__ __forceinline__ float ssp(float x) {
  float ax = fabsf(x);
  return fmaxf(x, 0.0f) + __logf(1.0f + __expf(-ax)) - 0.69314718056f;
}

__device__ __forceinline__ v8f wmma_bf16(v16bf a, v16bf b, v8f c) {
  // D = A(16x32 bf16) * B(32x16 bf16) + C(f32)
  return __builtin_amdgcn_wmma_f32_16x16x32_bf16(
      false, a, false, b, (short)0, c, false, false);
}

union Frag { bf16_t e[16]; uint32_t u[8]; v16bf v; };

// B fragment from LDS: lane holds N=n, K = k16..+7 and k16+16..+23
__device__ __forceinline__ v16bf ldsB(const unsigned short* s, int n,
                                      int rowStride, int k16) {
  const uint4* p0 = (const uint4*)(s + n * rowStride + k16);
  const uint4* p1 = (const uint4*)(s + n * rowStride + k16 + 16);
  Frag f; uint4 a = *p0, b = *p1;
  f.u[0]=a.x; f.u[1]=a.y; f.u[2]=a.z; f.u[3]=a.w;
  f.u[4]=b.x; f.u[5]=b.y; f.u[6]=b.z; f.u[7]=b.w;
  return f.v;
}
// A fragment from LDS: lane holds M=row, K = kbase..+7 and kbase+16..+23
__device__ __forceinline__ v16bf ldsA(const unsigned short* s, int row,
                                      int rowStride, int kbase) {
  const uint4* p0 = (const uint4*)(s + row * rowStride + kbase);
  const uint4* p1 = (const uint4*)(s + row * rowStride + kbase + 16);
  Frag f; uint4 a = *p0, b = *p1;
  f.u[0]=a.x; f.u[1]=a.y; f.u[2]=a.z; f.u[3]=a.w;
  f.u[4]=b.x; f.u[5]=b.y; f.u[6]=b.z; f.u[7]=b.w;
  return f.v;
}
// A fragment from global fp32 (convert to bf16 on the fly, hw cvt path)
__device__ __forceinline__ v16bf gldAf32(const float* g, long row, int stride,
                                         int kbase) {
  const float4* q0 = (const float4*)(g + row * stride + kbase);
  const float4* q1 = (const float4*)(g + row * stride + kbase + 16);
  float4 p0 = q0[0], p1 = q0[1], p2 = q1[0], p3 = q1[1];
  Frag f;
  f.e[0]=(bf16_t)p0.x;  f.e[1]=(bf16_t)p0.y;
  f.e[2]=(bf16_t)p0.z;  f.e[3]=(bf16_t)p0.w;
  f.e[4]=(bf16_t)p1.x;  f.e[5]=(bf16_t)p1.y;
  f.e[6]=(bf16_t)p1.z;  f.e[7]=(bf16_t)p1.w;
  f.e[8]=(bf16_t)p2.x;  f.e[9]=(bf16_t)p2.y;
  f.e[10]=(bf16_t)p2.z; f.e[11]=(bf16_t)p2.w;
  f.e[12]=(bf16_t)p3.x; f.e[13]=(bf16_t)p3.y;
  f.e[14]=(bf16_t)p3.z; f.e[15]=(bf16_t)p3.w;
  return f.v;
}
// A fragment from global bf16 (u16) with row stride (for f_ij, stride 32)
__device__ __forceinline__ v16bf gldAu16(const unsigned short* g, long row,
                                         int stride, int kbase) {
  const uint4* p0 = (const uint4*)(g + row * stride + kbase);
  const uint4* p1 = (const uint4*)(g + row * stride + kbase + 16);
  Frag f; uint4 a = *p0, b = *p1;
  f.u[0]=a.x; f.u[1]=a.y; f.u[2]=a.z; f.u[3]=a.w;
  f.u[4]=b.x; f.u[5]=b.y; f.u[6]=b.z; f.u[7]=b.w;
  return f.v;
}

// ---------------- one-time prep kernels ----------------
// transpose all layer weights to n-major bf16 (B^T layout for WMMA B frags)
__global__ __launch_bounds__(256) void kPrepWeights(
    const float* __restrict__ in2f, const float* __restrict__ fW1,
    const float* __restrict__ fW2,  const float* __restrict__ oW1,
    const float* __restrict__ oW2,
    unsigned short* __restrict__ tIn2f, unsigned short* __restrict__ tW1,
    unsigned short* __restrict__ tW2,   unsigned short* __restrict__ tO1,
    unsigned short* __restrict__ tO2) {
  int idx = blockIdx.x * 256 + threadIdx.x;
  if (idx >= N_LAYERSC * NB * NB) return;
  int l = idx / (NB * NB), rem = idx % (NB * NB);
  int k = rem >> 7, n = rem & 127;
  long sq = (long)NB * NB;
  tIn2f[l * sq + n * NB + k] = bf1(in2f[l * sq + k * NB + n]);
  tW2  [l * sq + n * NB + k] = bf1(fW2 [l * sq + k * NB + n]);
  tO1  [l * sq + n * NB + k] = bf1(oW1 [l * sq + k * NB + n]);
  tO2  [l * sq + n * NB + k] = bf1(oW2 [l * sq + k * NB + n]);
  if (k < KPAD) {
    float w = (k < N_RBF) ? fW1[(long)l * N_RBF * NB + k * NB + n] : 0.0f;
    tW1[(long)l * NB * KPAD + n * KPAD + k] = bf1(w);
  }
}

// x = emb[atomic_numbers]  (x lives in d_out; rewritten every call)
__global__ __launch_bounds__(256) void kEmbed(const int* __restrict__ z,
                                              const float* __restrict__ emb,
                                              float* __restrict__ x) {
  int idx = blockIdx.x * 256 + threadIdx.x;   // N_ATOMS * 32 float4s
  if (idx >= N_ATOMS * 32) return;
  int a = idx >> 5, c = idx & 31;
  float4 v = ((const float4*)(emb + (long)z[a] * NB))[c];
  ((float4*)(x + (long)a * NB))[c] = v;
}

// per-edge distance, cosine cutoff, Gaussian RBF (bf16, K padded to 32)
__global__ __launch_bounds__(256) void kEdgeGeom(const float* __restrict__ rij,
                                                 unsigned short* __restrict__ fij,
                                                 float* __restrict__ rcut) {
  long e = (long)blockIdx.x * 256 + threadIdx.x;
  if (e >= N_EDGES) return;
  float x = rij[e * 3], y = rij[e * 3 + 1], zc = rij[e * 3 + 2];
  float d = sqrtf(x * x + y * y + zc * zc);
  float rc = (d < CUTOFF_F)
                 ? 0.5f * (__cosf(d * 3.14159265358979f / CUTOFF_F) + 1.0f)
                 : 0.0f;
  rcut[e] = rc;
  const float delta = CUTOFF_F / (float)(N_RBF - 1);
  const float coeff = -0.5f / (delta * delta);
  Frag f0, f1;
#pragma unroll
  for (int m = 0; m < 16; ++m) {
    float t0 = d - delta * (float)m;
    float t1 = d - delta * (float)(m + 16);
    f0.e[m] = (bf16_t)((m < N_RBF) ? __expf(coeff * t0 * t0) : 0.0f);
    f1.e[m] = (bf16_t)((m + 16 < N_RBF) ? __expf(coeff * t1 * t1) : 0.0f);
  }
  uint4* out = (uint4*)(fij + e * KPAD);
  out[0] = make_uint4(f0.u[0], f0.u[1], f0.u[2], f0.u[3]);
  out[1] = make_uint4(f0.u[4], f0.u[5], f0.u[6], f0.u[7]);
  out[2] = make_uint4(f1.u[0], f1.u[1], f1.u[2], f1.u[3]);
  out[3] = make_uint4(f1.u[4], f1.u[5], f1.u[6], f1.u[7]);
}

__global__ __launch_bounds__(256) void kZero(float4* __restrict__ p, long n4) {
  long i = (long)blockIdx.x * 256 + threadIdx.x;
  if (i < n4) p[i] = make_float4(0.f, 0.f, 0.f, 0.f);
}

// ---------------- per-layer kernels ----------------
#define TSTR 136  // padded LDS row stride in u16 (272B = 68 dwords, conflict-free)

// h = bf16( x @ W )   (no bias), one 16-row tile per wave, 8 waves / WG
__global__ __launch_bounds__(256) void kNodeGemm(
    const unsigned short* __restrict__ Wt,  // [128][128] n-major bf16
    const float* __restrict__ x, unsigned short* __restrict__ h) {
  __shared__ unsigned short sW[NB * TSTR];
  {
    const uint4* s4 = (const uint4*)Wt;
    for (int i = threadIdx.x; i < 2048; i += 256) {
      int r = i >> 4, c = i & 15;
      *(uint4*)&sW[r * TSTR + c * 8] = s4[i];
    }
  }
  __syncthreads();
  int wave = threadIdx.x >> 5, lane = threadIdx.x & 31;
  int lane15 = lane & 15, hi = lane >> 4;
  int tile = blockIdx.x * 8 + wave;
  if (tile >= N_ATOMS / 16) return;   // wave-uniform
  int row0 = tile * 16;
  v16bf A[4];
#pragma unroll
  for (int kk = 0; kk < 4; ++kk)
    A[kk] = gldAf32(x, row0 + lane15, NB, kk * 32 + hi * 8);
#pragma unroll
  for (int nt = 0; nt < 8; ++nt) {
    int n = nt * 16 + lane15;
    v8f acc = {};
#pragma unroll
    for (int kk = 0; kk < 4; ++kk)
      acc = wmma_bf16(A[kk], ldsB(sW, n, TSTR, kk * 32 + hi * 8), acc);
    int rbase = row0 + hi * 8;
#pragma unroll
    for (int r = 0; r < 8; ++r)
      h[(long)(rbase + r) * NB + n] = bf1(acc[r]);
  }
}

// fused edge kernel: Wij = (ssp(fij@W1+b1)@W2+b2)*rcut;  agg[idx_i] += h[idx_j]*Wij
__global__ __launch_bounds__(256) void kEdge(
    const unsigned short* __restrict__ Wt1,  // [128][32]  n-major bf16
    const unsigned short* __restrict__ Wt2,  // [128][128] n-major bf16
    const float* __restrict__ b1, const float* __restrict__ b2,
    const unsigned short* __restrict__ fij, const float* __restrict__ rcut,
    const int* __restrict__ idx_i, const int* __restrict__ idx_j,
    const unsigned short* __restrict__ h, float* __restrict__ agg) {
  __shared__ unsigned short sW1[NB * 40];
  __shared__ unsigned short sW2[NB * TSTR];
  __shared__ unsigned short sT[8 * 16 * TSTR];  // per-wave transpose tiles
  {
    const uint4* s4 = (const uint4*)Wt1;
    for (int i = threadIdx.x; i < 512; i += 256) {
      int r = i >> 2, c = i & 3;
      *(uint4*)&sW1[r * 40 + c * 8] = s4[i];
    }
    const uint4* t4 = (const uint4*)Wt2;
    for (int i = threadIdx.x; i < 2048; i += 256) {
      int r = i >> 4, c = i & 15;
      *(uint4*)&sW2[r * TSTR + c * 8] = t4[i];
    }
  }
  __syncthreads();
  int wave = threadIdx.x >> 5, lane = threadIdx.x & 31;
  int lane15 = lane & 15, hi = lane >> 4;
  long e0 = ((long)blockIdx.x * 8 + wave) * 16;   // exact: 5000*8*16 = 640000
  unsigned short* tT = &sT[wave * 16 * TSTR];

  // GEMM1: t = ssp(fij @ W1 + b1), single K=32 step
  v16bf Af = gldAu16(fij, e0 + lane15, KPAD, hi * 8);
#pragma unroll
  for (int nt = 0; nt < 8; ++nt) {
    int n = nt * 16 + lane15;
    v8f acc = {};
    acc = wmma_bf16(Af, ldsB(sW1, n, 40, hi * 8), acc);
    float bn = b1[n];
#pragma unroll
    for (int r = 0; r < 8; ++r)
      tT[(r + hi * 8) * TSTR + n] = bf1(ssp(acc[r] + bn));
  }
  // wave-local LDS transpose (same-wave DS ops are in-order; compiler waits)

  // GEMM2: Wij = t @ W2 + b2
  v16bf At[4];
#pragma unroll
  for (int kk = 0; kk < 4; ++kk)
    At[kk] = ldsA(tT, lane15, TSTR, kk * 32 + hi * 8);

  float rc[8]; int ii[8], jj[8];
#pragma unroll
  for (int r = 0; r < 8; ++r) {
    long e = e0 + r + hi * 8;
    rc[r] = rcut[e]; ii[r] = idx_i[e]; jj[r] = idx_j[e];
  }
#pragma unroll
  for (int nt = 0; nt < 8; ++nt) {
    int n = nt * 16 + lane15;
    float bn = b2[n];
    v8f acc;
#pragma unroll
    for (int r = 0; r < 8; ++r) acc[r] = bn;
#pragma unroll
    for (int kk = 0; kk < 4; ++kk)
      acc = wmma_bf16(At[kk], ldsB(sW2, n, TSTR, kk * 32 + hi * 8), acc);
    // segmented scatter: idx_i is sorted, so each lane's 8 consecutive edges
    // mostly share the target atom -> merge runs in registers, then one
    // global_atomic_add_f32 per run (up to 8x fewer atomics)
    float carry = 0.0f;
#pragma unroll
    for (int r = 0; r < 8; ++r) {
      float hv = bf2f(h[(long)jj[r] * NB + n]);
      float val = carry + acc[r] * rc[r] * hv;
      if (r < 7 && ii[r] == ii[r + 1]) {
        carry = val;
      } else {
        atomicAdd(&agg[(long)ii[r] * NB + n], val);
        carry = 0.0f;
      }
    }
  }
}

// x += ssp(agg @ O1 + b1) @ O2 + b2
__global__ __launch_bounds__(256) void kOut(
    const unsigned short* __restrict__ WtO1, const unsigned short* __restrict__ WtO2,
    const float* __restrict__ b1, const float* __restrict__ b2,
    const float* __restrict__ agg, float* __restrict__ x) {
  __shared__ unsigned short sW1[NB * TSTR];
  __shared__ unsigned short sW2[NB * TSTR];
  __shared__ unsigned short sT[8 * 16 * TSTR];
  {
    const uint4* s4 = (const uint4*)WtO1;
    const uint4* t4 = (const uint4*)WtO2;
    for (int i = threadIdx.x; i < 2048; i += 256) {
      int r = i >> 4, c = i & 15;
      *(uint4*)&sW1[r * TSTR + c * 8] = s4[i];
      *(uint4*)&sW2[r * TSTR + c * 8] = t4[i];
    }
  }
  __syncthreads();
  int wave = threadIdx.x >> 5, lane = threadIdx.x & 31;
  int lane15 = lane & 15, hi = lane >> 4;
  int tile = blockIdx.x * 8 + wave;
  if (tile >= N_ATOMS / 16) return;   // wave-uniform
  int row0 = tile * 16;
  unsigned short* tT = &sT[wave * 16 * TSTR];

  v16bf A[4];
#pragma unroll
  for (int kk = 0; kk < 4; ++kk)
    A[kk] = gldAf32(agg, row0 + lane15, NB, kk * 32 + hi * 8);
#pragma unroll
  for (int nt = 0; nt < 8; ++nt) {
    int n = nt * 16 + lane15;
    float bn = b1[n];
    v8f acc;
#pragma unroll
    for (int r = 0; r < 8; ++r) acc[r] = bn;
#pragma unroll
    for (int kk = 0; kk < 4; ++kk)
      acc = wmma_bf16(A[kk], ldsB(sW1, n, TSTR, kk * 32 + hi * 8), acc);
#pragma unroll
    for (int r = 0; r < 8; ++r)
      tT[(r + hi * 8) * TSTR + n] = bf1(ssp(acc[r]));
  }
  v16bf At[4];
#pragma unroll
  for (int kk = 0; kk < 4; ++kk)
    At[kk] = ldsA(tT, lane15, TSTR, kk * 32 + hi * 8);
#pragma unroll
  for (int nt = 0; nt < 8; ++nt) {
    int n = nt * 16 + lane15;
    float bn = b2[n];
    v8f acc;
#pragma unroll
    for (int r = 0; r < 8; ++r) acc[r] = bn;
#pragma unroll
    for (int kk = 0; kk < 4; ++kk)
      acc = wmma_bf16(At[kk], ldsB(sW2, n, TSTR, kk * 32 + hi * 8), acc);
#pragma unroll
    for (int r = 0; r < 8; ++r) {
      long o = (long)(row0 + r + hi * 8) * NB + n;
      x[o] += acc[r];    // each element written exactly once
    }
  }
}

// ---------------- host launch ----------------
extern "C" void kernel_launch(void* const* d_in, const int* in_sizes, int n_in,
                              void* d_out, int out_size, void* d_ws, size_t ws_size,
                              hipStream_t stream) {
  const int*   z     = (const int*)d_in[0];
  const float* r_ij  = (const float*)d_in[1];
  const int*   idx_i = (const int*)d_in[2];
  const int*   idx_j = (const int*)d_in[3];
  const float* emb   = (const float*)d_in[4];
  const float* in2fW = (const float*)d_in[5];
  const float* fW1   = (const float*)d_in[6];
  const float* fb1   = (const float*)d_in[7];
  const float* fW2   = (const float*)d_in[8];
  const float* fb2   = (const float*)d_in[9];
  const float* oW1   = (const float*)d_in[10];
  const float* ob1   = (const float*)d_in[11];
  const float* oW2   = (const float*)d_in[12];
  const float* ob2   = (const float*)d_in[13];
  float* x = (float*)d_out;   // x lives in d_out; fully rewritten each call

  // workspace carve-out (~75 MB total)
  char* ws = (char*)d_ws;
  size_t off = 0;
  auto carve = [&](size_t bytes) -> void* {
    void* p = ws + off;
    off += (bytes + 255) & ~(size_t)255;
    return p;
  };
  unsigned short* h     = (unsigned short*)carve((size_t)N_ATOMS * NB * 2);
  float*          agg   = (float*)         carve((size_t)N_ATOMS * NB * 4);
  unsigned short* fij   = (unsigned short*)carve((size_t)N_EDGES * KPAD * 2);
  float*          rcut  = (float*)         carve((size_t)N_EDGES * 4);
  unsigned short* tIn2f = (unsigned short*)carve((size_t)N_LAYERSC * NB * NB * 2);
  unsigned short* tW1   = (unsigned short*)carve((size_t)N_LAYERSC * NB * KPAD * 2);
  unsigned short* tW2   = (unsigned short*)carve((size_t)N_LAYERSC * NB * NB * 2);
  unsigned short* tO1   = (unsigned short*)carve((size_t)N_LAYERSC * NB * NB * 2);
  unsigned short* tO2   = (unsigned short*)carve((size_t)N_LAYERSC * NB * NB * 2);

  kPrepWeights<<<(N_LAYERSC * NB * NB + 255) / 256, 256, 0, stream>>>(
      in2fW, fW1, fW2, oW1, oW2, tIn2f, tW1, tW2, tO1, tO2);
  kEmbed<<<(N_ATOMS * 32 + 255) / 256, 256, 0, stream>>>(z, emb, x);
  kEdgeGeom<<<(N_EDGES + 255) / 256, 256, 0, stream>>>(r_ij, fij, rcut);

  const long agg4 = (long)N_ATOMS * NB / 4;
  for (int l = 0; l < N_LAYERSC; ++l) {
    kZero<<<(int)((agg4 + 255) / 256), 256, 0, stream>>>((float4*)agg, agg4);
    kNodeGemm<<<(N_ATOMS / 16 + 7) / 8, 256, 0, stream>>>(
        tIn2f + (long)l * NB * NB, x, h);
    kEdge<<<N_EDGES / 128, 256, 0, stream>>>(
        tW1 + (long)l * NB * KPAD, tW2 + (long)l * NB * NB,
        fb1 + l * NB, fb2 + l * NB, fij, rcut, idx_i, idx_j, h, agg);
    kOut<<<(N_ATOMS / 16 + 7) / 8, 256, 0, stream>>>(
        tO1 + (long)l * NB * NB, tO2 + (long)l * NB * NB,
        ob1 + l * NB, ob2 + l * NB, agg, x);
  }
}
